// MoeFusedLinear_403726926533
// MI455X (gfx1250) — matrix operand order repaired and checked by reading, not compiled, not run on hardware
//
#include <hip/hip_runtime.h>

// ---------------------------------------------------------------------------
// MoE grouped GEMM for MI455X (gfx1250, wave32).
// out[m, n] = sum_k x[m, k] * w[expert(m), n, k]
// fp32 -> bf16 staged in LDS, V_WMMA_F32_16X16X32_BF16, fp32 accumulate.
// 128x128 block tile, BK=32, double-buffered LDS, 1 barrier / K-step.
// sched_barrier(0) fences split each iteration into: [issue 8 global b128
// prefetches] [12 ds_load_b128 + 8 wmma] [cvt + ds_store], so global latency
// is covered by the whole compute block instead of stalling mid-iteration.
// ---------------------------------------------------------------------------

typedef __attribute__((ext_vector_type(16))) __bf16 bf16x16;
typedef __attribute__((ext_vector_type(8)))  float  v8f;

#define NEXP      8
#define K_FEAT    2048
#define N_FEAT    2048
#define BM        128
#define BN        128
#define BK        32
#define LDA       40      // padded LDS row stride (elements): conflict-free b128
#define LDB       40
#define NTHREADS  256     // 8 waves
#define NKSTEPS   (K_FEAT / BK)

union FragAB { bf16x16 v; uint4 u[2]; };
union Pack4  { __bf16 h[4]; uint2 u; };

static __device__ __forceinline__ __bf16 f2bf(float f) { return (__bf16)f; }

__global__ __launch_bounds__(NTHREADS) void moe_grouped_gemm_bf16(
    const float* __restrict__ x,        // [M_TOTAL, K]
    const float* __restrict__ w,        // [E, N, K]
    const int*   __restrict__ m_sizes,  // [E]
    float*       __restrict__ out)      // [M_TOTAL, N]
{
    __shared__ __bf16 As[2][BM * LDA];
    __shared__ __bf16 Bs[2][BN * LDB];

    // ---- map blockIdx.y -> (expert, m-tile). Tiles never cross experts. ----
    int e = -1, m0 = 0, rows = 0;
    {
        const int by = (int)blockIdx.y;
        int tile_acc = 0, row_base = 0;
        #pragma unroll
        for (int i = 0; i < NEXP; ++i) {
            const int me = m_sizes[i];                 // uniform load
            const int t  = (me + BM - 1) / BM;
            if (e < 0 && by < tile_acc + t) {
                const int tile = by - tile_acc;
                e    = i;
                m0   = row_base + tile * BM;
                const int rem = me - tile * BM;
                rows = rem < BM ? rem : BM;
            }
            tile_acc += t;
            row_base += me;
        }
        if (e < 0) return;                             // past last real tile
    }

    const int tid  = (int)threadIdx.x;
    const int lane = tid & 31;
    const int wv   = tid >> 5;
    const int wm   = wv >> 2;                          // 0..1  -> 64-row slab
    const int wn   = wv & 3;                           // 0..3  -> 32-col slab
    const int n0   = (int)blockIdx.x * BN;

    const float* __restrict__ wexp = w + (size_t)e * N_FEAT * K_FEAT;

    // ---- per-thread staging addresses, hoisted out of the K loop ----------
    const float* aPtr[4];
    const float* bPtr[4];
    int          sRow[4], sCol[4];
    #pragma unroll
    for (int i = 0; i < 4; ++i) {
        const int idx = tid + i * NTHREADS;            // 0..1023
        const int r   = idx >> 3;                      // 0..127
        const int c   = (idx & 7) << 2;                // 0,4,...,28
        sRow[i] = r; sCol[i] = c;
        // clamp OOB rows of a partial expert tile to row 0 (in-bounds; data
        // is never stored because the epilogue guards on `rows`)
        const int ra  = (r < rows) ? r : 0;
        aPtr[i] = x    + (size_t)(m0 + ra) * K_FEAT + c;
        bPtr[i] = wexp + (size_t)(n0 + r)  * K_FEAT + c;
    }

    float4 aReg[4], bReg[4];

    auto gload = [&](int kt) {
        const int k0 = kt * BK;
        #pragma unroll
        for (int i = 0; i < 4; ++i) {
            aReg[i] = *(const float4*)(aPtr[i] + k0);  // global_load_b128
            bReg[i] = *(const float4*)(bPtr[i] + k0);
        }
    };

    auto lstore = [&](int buf) {
        #pragma unroll
        for (int i = 0; i < 4; ++i) {
            Pack4 pa, pb;
            pa.h[0] = f2bf(aReg[i].x); pa.h[1] = f2bf(aReg[i].y);
            pa.h[2] = f2bf(aReg[i].z); pa.h[3] = f2bf(aReg[i].w);
            pb.h[0] = f2bf(bReg[i].x); pb.h[1] = f2bf(bReg[i].y);
            pb.h[2] = f2bf(bReg[i].z); pb.h[3] = f2bf(bReg[i].w);
            *(uint2*)&As[buf][sRow[i] * LDA + sCol[i]] = pa.u;   // ds_store_b64
            *(uint2*)&Bs[buf][sRow[i] * LDB + sCol[i]] = pb.u;
        }
    };

    v8f acc[4][2] = {};                                // 64 f32 accum VGPRs

    const int fr = lane & 15;                          // row/col within fragment
    const int fh = lane >> 4;                          // half-wave selector

    auto compute = [&](int buf) {
        FragAB af[4], bfr[2];
        // Load B fragments first, then A in fm order: the fm-major wmma
        // sequence then only depends on early loads -> graduated dscnt waits.
        // B 32x16 bf16 layout: lane<16 holds K{0..7,16..23} of col fr, lane>=16 K{8..15,24..31}
        #pragma unroll
        for (int fn = 0; fn < 2; ++fn) {
            const int col = wn * 32 + fn * 16 + fr;
            const __bf16* p = &Bs[buf][col * LDB + fh * 8];
            bfr[fn].u[0] = *(const uint4*)p;           // ds_load_b128
            bfr[fn].u[1] = *(const uint4*)(p + 16);
        }
        // A 16x32 bf16 layout: lane<16 holds K{0..7,16..23}, lane>=16 K{8..15,24..31}
        #pragma unroll
        for (int fm = 0; fm < 4; ++fm) {
            const int row = wm * 64 + fm * 16 + fr;
            const __bf16* p = &As[buf][row * LDA + fh * 8];
            af[fm].u[0] = *(const uint4*)p;
            af[fm].u[1] = *(const uint4*)(p + 16);
        }
        #pragma unroll
        for (int fm = 0; fm < 4; ++fm)
            #pragma unroll
            for (int fn = 0; fn < 2; ++fn)
                acc[fm][fn] = __builtin_amdgcn_wmma_f32_16x16x32_bf16(
                    false, af[fm].v, false, bfr[fn].v,
                    (short)0, acc[fm][fn], false, false);
    };

    // ---- main K loop: fence scheduling regions so all 8 global loads issue
    //      at the top and their waits/cvts sit below the whole wmma block ----
    gload(0);
    lstore(0);
    __syncthreads();
    for (int kt = 0; kt < NKSTEPS; ++kt) {
        const int cur = kt & 1;
        if (kt + 1 < NKSTEPS) gload(kt + 1);
        __builtin_amdgcn_sched_barrier(0);             // loads stay above
        compute(cur);
        __builtin_amdgcn_sched_barrier(0);             // cvt/stores stay below
        if (kt + 1 < NKSTEPS) lstore(cur ^ 1);
        __syncthreads();
    }

    // ---- epilogue: C/D 16x16 f32 layout: VGPR v -> M = v + 8*(lane>=16) ----
    #pragma unroll
    for (int fm = 0; fm < 4; ++fm) {
        #pragma unroll
        for (int fn = 0; fn < 2; ++fn) {
            const int col = n0 + wn * 32 + fn * 16 + fr;
            #pragma unroll
            for (int v = 0; v < 8; ++v) {
                const int rr = wm * 64 + fm * 16 + fh * 8 + v;
                if (rr < rows)
                    out[(size_t)(m0 + rr) * N_FEAT + col] = acc[fm][fn][v];
            }
        }
    }
}

extern "C" void kernel_launch(void* const* d_in, const int* in_sizes, int n_in,
                              void* d_out, int out_size, void* d_ws, size_t ws_size,
                              hipStream_t stream) {
    const float* x  = (const float*)d_in[0];
    const float* w  = (const float*)d_in[1];
    const int*   ms = (const int*)d_in[2];
    float*       o  = (float*)d_out;

    const int M = in_sizes[0] / K_FEAT;                // 16384
    // worst case: sum_e ceil(m_e/BM) <= M/BM + E - 1
    dim3 grid(N_FEAT / BN, M / BM + NEXP - 1);
    moe_grouped_gemm_bf16<<<grid, NTHREADS, 0, stream>>>(x, w, ms, o);
}
